// Attention_85864986182156
// MI455X (gfx1250) — matrix operand order repaired and checked by reading, not compiled, not run on hardware
//
#include <hip/hip_runtime.h>
#include <hip/hip_bf16.h>

// ---------------------------------------------------------------------------
// MI455X / gfx1250 fused attention:
//   1) fp32 -> bf16 convert of x, w_qkv, w_out
//   2) QKV GEMM   (bf16 WMMA, fp32 accum)  -> qkv buffer [8192 x 3072] bf16
//   3) Flash attention per (b,h,qtile)     -> attn out  [8192 x 1024] bf16
//   4) Output GEMM (bf16 WMMA, fp32 accum) -> d_out fp32
// Workspace layout (bf16 elements), total ~88 MB:
//   xb[8192*1024] | wqkvb[3072*1024] | wob[1024*1024] | qkvb[8192*3072] | aob[8192*1024]
// ---------------------------------------------------------------------------

typedef __attribute__((ext_vector_type(16))) __bf16 v16bf;
typedef __attribute__((ext_vector_type(8)))  __bf16 v8bf;
typedef __attribute__((ext_vector_type(4)))  __bf16 v4bf;
typedef __attribute__((ext_vector_type(8)))  float  v8f;

#define WMMA_BF16(a, b, c) \
  __builtin_amdgcn_wmma_f32_16x16x32_bf16(false, (a), false, (b), (short)0, (c), false, false)

// A-fragment (16x32 bf16): lane holds M = lane%16; element e -> K:
//   e=0..15 : K = half*16 + e          (32B contiguous)
static __device__ __forceinline__ v16bf frag_a(const __bf16* row, int kbase, int half) {
  union { v16bf v; v8bf h[2]; } u;
  u.h[0] = *(const v8bf*)(row + kbase + half * 16);
  u.h[1] = *(const v8bf*)(row + kbase + half * 16 + 8);
  return u.v;
}

// ---------------------------------------------------------------------------
// Kernel 1: fp32 -> bf16 convert, 4-wide
// ---------------------------------------------------------------------------
__global__ void cvt_f32_bf16(const float* __restrict__ in, __bf16* __restrict__ out, int n4) {
  int i = blockIdx.x * blockDim.x + threadIdx.x;
  int stride = gridDim.x * blockDim.x;
  for (; i < n4; i += stride) {
    float4 f = ((const float4*)in)[i];
    v4bf o;
    o[0] = (__bf16)f.x; o[1] = (__bf16)f.y; o[2] = (__bf16)f.z; o[3] = (__bf16)f.w;
    ((v4bf*)out)[i] = o;
  }
}

// ---------------------------------------------------------------------------
// Kernel 2/4: C[M,N] = A[M,K] * B[N,K]^T   (both K-major, bf16, fp32 accum)
// 256 threads = 8 waves (2x4). Block tile 128x128, wave tile 64x32 (4x2 WMMA).
// ---------------------------------------------------------------------------
template <bool STORE_BF16>
__global__ __launch_bounds__(256)
void gemm_nt(const __bf16* __restrict__ A, const __bf16* __restrict__ B,
             float* __restrict__ Cf, __bf16* __restrict__ Cb,
             int M, int N, int K) {
  __shared__ __bf16 As[128][48];   // 32-wide K slice, stride 48 (96B, 32B-aligned rows)
  __shared__ __bf16 Bs[128][48];

  const int tid  = threadIdx.x;
  const int lane = tid & 31;
  const int wave = tid >> 5;
  const int half = lane >> 4;
  const int lm   = lane & 15;
  const int wm   = wave >> 2;           // 0..1
  const int wn   = wave & 3;            // 0..3
  const int bm   = blockIdx.y * 128;
  const int bn   = blockIdx.x * 128;

  const int lrow = tid >> 1;            // 0..127
  const int lcol = (tid & 1) * 16;      // 0 or 16
  const __bf16* gA = A + (size_t)(bm + lrow) * K + lcol;
  const __bf16* gB = B + (size_t)(bn + lrow) * K + lcol;

  v8f acc[4][2];
#pragma unroll
  for (int mi = 0; mi < 4; ++mi)
#pragma unroll
    for (int ni = 0; ni < 2; ++ni) acc[mi][ni] = v8f{};

  for (int k0 = 0; k0 < K; k0 += 32) {
    *(v16bf*)&As[lrow][lcol] = *(const v16bf*)(gA + k0);
    *(v16bf*)&Bs[lrow][lcol] = *(const v16bf*)(gB + k0);
    if (k0 + 32 < K) {                      // global_prefetch_b8 of next K slice
      __builtin_prefetch(gA + k0 + 32, 0, 0);
      __builtin_prefetch(gB + k0 + 32, 0, 0);
    }
    __syncthreads();

    v16bf af[4], bfr[2];
#pragma unroll
    for (int mi = 0; mi < 4; ++mi)
      af[mi] = frag_a(&As[wm * 64 + mi * 16 + lm][0], 0, half);
#pragma unroll
    for (int ni = 0; ni < 2; ++ni)          // B-frag: N=lm, K = 16*half + e contiguous
      bfr[ni] = *(const v16bf*)(&Bs[wn * 32 + ni * 16 + lm][half * 16]);

#pragma unroll
    for (int mi = 0; mi < 4; ++mi)
#pragma unroll
      for (int ni = 0; ni < 2; ++ni)
        acc[mi][ni] = WMMA_BF16(af[mi], bfr[ni], acc[mi][ni]);

    __syncthreads();
  }

#pragma unroll
  for (int mi = 0; mi < 4; ++mi)
#pragma unroll
    for (int ni = 0; ni < 2; ++ni) {
      const int col = bn + wn * 32 + ni * 16 + lm;
#pragma unroll
      for (int r = 0; r < 8; ++r) {
        const int row = bm + wm * 64 + mi * 16 + half * 8 + r;
        if constexpr (STORE_BF16)
          Cb[(size_t)row * N + col] = (__bf16)acc[mi][ni][r];
        else
          Cf[(size_t)row * N + col] = acc[mi][ni][r];
      }
    }
}

// ---------------------------------------------------------------------------
// Kernel 3: flash attention, one workgroup per (qtile=64 rows, head, batch).
// 128 threads = 4 waves; each wave owns 16 q rows. Streams 64-key tiles.
// qkv layout: [b*2048+n][3072] with q at +0, k at +1024, v at +2048, head h at +h*64.
// ---------------------------------------------------------------------------
__global__ __launch_bounds__(128)
void flash_attn(const __bf16* __restrict__ qkv, __bf16* __restrict__ out) {
  __shared__ __bf16 Ks[64][80];        // keys x d, row-major
  __shared__ __bf16 Vt[64][80];        // d x keys (transposed at load)
  __shared__ __bf16 Ps[4][16][80];     // per-wave P tile (16 q x 64 keys)

  constexpr float kScale = 0.125f;     // 1/sqrt(64)
  const int tid  = threadIdx.x;
  const int lane = tid & 31;
  const int wave = tid >> 5;
  const int half = lane >> 4;
  const int lm   = lane & 15;
  const int h    = blockIdx.y;
  const int b    = blockIdx.z;
  const int q0   = blockIdx.x * 64 + wave * 16;

  const size_t tok = 3072;
  const __bf16* qbase = qkv + (size_t)(b * 2048) * tok + h * 64;
  const __bf16* kbase = qbase + 1024;
  const __bf16* vbase = qbase + 2048;

  // Resident Q fragments (16 rows x 64 d => two 16x32 A-fragments)
  v16bf aq[2];
  {
    const __bf16* qrow = qbase + (size_t)(q0 + lm) * tok;
    aq[0] = frag_a(qrow, 0, half);
    aq[1] = frag_a(qrow, 32, half);
  }

  v8f o[4];
#pragma unroll
  for (int d = 0; d < 4; ++d) o[d] = v8f{};
  float mrow[8], lrow[8];
#pragma unroll
  for (int r = 0; r < 8; ++r) { mrow[r] = -3.0e38f; lrow[r] = 0.0f; }

  const int lrd = tid >> 1;            // 0..63  (tile row for cooperative loads)
  const int lcd = (tid & 1) * 32;      // 0 or 32

  for (int kt = 0; kt < 32; ++kt) {
    const int key0 = kt * 64;

    // --- stage K tile (row-major) ---
    const __bf16* kp = kbase + (size_t)(key0 + lrd) * tok + lcd;
    *(v16bf*)&Ks[lrd][lcd]      = *(const v16bf*)(kp);
    *(v16bf*)&Ks[lrd][lcd + 16] = *(const v16bf*)(kp + 16);

    // --- stage V tile transposed: Vt[d][key] ---
    const __bf16* vp = vbase + (size_t)(key0 + lrd) * tok + lcd;
    union { v16bf v; __bf16 e[16]; } u0, u1;
    u0.v = *(const v16bf*)(vp);
    u1.v = *(const v16bf*)(vp + 16);
#pragma unroll
    for (int i = 0; i < 16; ++i) {
      Vt[lcd + i][lrd]      = u0.e[i];
      Vt[lcd + 16 + i][lrd] = u1.e[i];
    }
    __syncthreads();

    // --- S = Q K^T for 4 key sub-tiles of 16 ---
    v8f s[4];
#pragma unroll
    for (int j = 0; j < 4; ++j) {
      v8f z{};
      v16bf b0 = *(const v16bf*)(&Ks[j * 16 + lm][half * 16]);       // d in [0,32)
      v16bf b1 = *(const v16bf*)(&Ks[j * 16 + lm][32 + half * 16]);  // d in [32,64)
      s[j] = WMMA_BF16(aq[0], b0, z);
      s[j] = WMMA_BF16(aq[1], b1, s[j]);
    }

    // --- online softmax (row M = half*8 + r lives in element r, across 16 lanes) ---
#pragma unroll
    for (int r = 0; r < 8; ++r) {
      float sv[4];
#pragma unroll
      for (int j = 0; j < 4; ++j) sv[j] = s[j][r] * kScale;
      float v = fmaxf(fmaxf(sv[0], sv[1]), fmaxf(sv[2], sv[3]));
#pragma unroll
      for (int off = 8; off >= 1; off >>= 1) v = fmaxf(v, __shfl_xor(v, off, 32));
      const float mnew = fmaxf(mrow[r], v);
      const float sc   = __expf(mrow[r] - mnew);
      mrow[r] = mnew;
      float sum = 0.0f;
#pragma unroll
      for (int j = 0; j < 4; ++j) {
        const float p = __expf(sv[j] - mnew);
        Ps[wave][half * 8 + r][j * 16 + lm] = (__bf16)p;
        sum += p;
      }
#pragma unroll
      for (int off = 8; off >= 1; off >>= 1) sum += __shfl_xor(sum, off, 32);
      lrow[r] = lrow[r] * sc + sum;
#pragma unroll
      for (int d = 0; d < 4; ++d) o[d][r] *= sc;
    }
    // in-wave LDS RAW: P stores -> P fragment loads
    asm volatile("s_wait_dscnt 0x0" ::: "memory");

    // --- O += P V ---
#pragma unroll
    for (int d = 0; d < 4; ++d) {
#pragma unroll
      for (int ks = 0; ks < 2; ++ks) {
        v16bf ap = frag_a(&Ps[wave][lm][0], ks * 32, half);
        v16bf bv = *(const v16bf*)(&Vt[d * 16 + lm][ks * 32 + half * 16]);
        o[d] = WMMA_BF16(ap, bv, o[d]);
      }
    }
    __syncthreads();
  }

  // --- normalize and store bf16 (merged-head layout [row][h*64+d]) ---
#pragma unroll
  for (int r = 0; r < 8; ++r) {
    const float inv = 1.0f / lrow[r];
    const int row = b * 2048 + q0 + half * 8 + r;
#pragma unroll
    for (int d = 0; d < 4; ++d)
      out[(size_t)row * 1024 + h * 64 + d * 16 + lm] = (__bf16)(o[d][r] * inv);
  }
}

// ---------------------------------------------------------------------------
extern "C" void kernel_launch(void* const* d_in, const int* in_sizes, int n_in,
                              void* d_out, int out_size, void* d_ws, size_t ws_size,
                              hipStream_t stream) {
  (void)in_sizes; (void)n_in; (void)out_size; (void)ws_size;

  const float* x     = (const float*)d_in[0];
  const float* w_qkv = (const float*)d_in[1];
  const float* w_out = (const float*)d_in[2];
  float* out = (float*)d_out;

  __bf16* xb    = (__bf16*)d_ws;
  __bf16* wqkvb = xb    + (size_t)8192 * 1024;
  __bf16* wob   = wqkvb + (size_t)3072 * 1024;
  __bf16* qkvb  = wob   + (size_t)1024 * 1024;
  __bf16* aob   = qkvb  + (size_t)8192 * 3072;

  int n4;
  n4 = (8192 * 1024) / 4;
  cvt_f32_bf16<<<(n4 + 255) / 256, 256, 0, stream>>>(x, xb, n4);
  n4 = (3072 * 1024) / 4;
  cvt_f32_bf16<<<(n4 + 255) / 256, 256, 0, stream>>>(w_qkv, wqkvb, n4);
  n4 = (1024 * 1024) / 4;
  cvt_f32_bf16<<<(n4 + 255) / 256, 256, 0, stream>>>(w_out, wob, n4);

  // qkv = x @ w_qkv^T : M=8192, N=3072, K=1024 -> bf16
  gemm_nt<true><<<dim3(3072 / 128, 8192 / 128), 256, 0, stream>>>(
      xb, wqkvb, nullptr, qkvb, 8192, 3072, 1024);

  // flash attention: grid (qtiles=32, heads=16, batch=4)
  flash_attn<<<dim3(32, 16, 4), 128, 0, stream>>>(qkvb, aob);

  // out = attn @ w_out^T : M=8192, N=1024, K=1024 -> fp32
  gemm_nt<false><<<dim3(1024 / 128, 8192 / 128), 256, 0, stream>>>(
      aob, wob, out, nullptr, 8192, 1024, 1024);
}
